// Attention_73212012527806
// MI455X (gfx1250) — matrix operand order repaired and checked by reading, not compiled, not run on hardware
//
#include <hip/hip_runtime.h>
#include <hip/hip_bf16.h>

#define NDIM 4096
#define BM 128
#define BN 128
#define BK 64
#define LDS_STRIDE 80   // bf16 elements per LDS row (64 + 16 pad) = 160 bytes

typedef __attribute__((ext_vector_type(16))) __bf16 v16bf;
typedef __attribute__((ext_vector_type(8)))  __bf16 v8bf;
typedef __attribute__((ext_vector_type(8)))  float  v8f;

__device__ __forceinline__ unsigned int f32_to_bf16(float f) {
  unsigned int u = __builtin_bit_cast(unsigned int, f);
  u += 0x7FFFu + ((u >> 16) & 1u);          // round to nearest even
  return (u >> 16) & 0xFFFFu;
}

__device__ __forceinline__ uint4 pack_bf16x8(float4 a, float4 b) {
  uint4 r;
  r.x = f32_to_bf16(a.x) | (f32_to_bf16(a.y) << 16);
  r.y = f32_to_bf16(a.z) | (f32_to_bf16(a.w) << 16);
  r.z = f32_to_bf16(b.x) | (f32_to_bf16(b.y) << 16);
  r.w = f32_to_bf16(b.z) | (f32_to_bf16(b.w) << 16);
  return r;
}

// Async copy of 64 contiguous bytes (32 bf16) global -> LDS, tracked by ASYNCcnt.
// INST_OFFSET is added to both the LDS and global address (ISA 08_async_tensor §4.4),
// so one LDS-offset VGPR + one 64-bit address pair serves all four b128 chunks.
// Low 32 bits of a generic pointer to __shared__ data are the LDS byte offset
// (flat-aperture rule).
__device__ __forceinline__ void async_copy_row64B(const unsigned short* gsrc, __bf16* ldst) {
  const unsigned lds = (unsigned)(size_t)(void*)ldst;
  const unsigned long long ga = (unsigned long long)(size_t)gsrc;
  asm volatile("global_load_async_to_lds_b128 %0, %1, off"           :: "v"(lds), "v"(ga) : "memory");
  asm volatile("global_load_async_to_lds_b128 %0, %1, off offset:16" :: "v"(lds), "v"(ga) : "memory");
  asm volatile("global_load_async_to_lds_b128 %0, %1, off offset:32" :: "v"(lds), "v"(ga) : "memory");
  asm volatile("global_load_async_to_lds_b128 %0, %1, off offset:48" :: "v"(lds), "v"(ga) : "memory");
}

// Elementwise fp32 -> bf16 (RNE), 16 elements/thread, fully coalesced.
__global__ __launch_bounds__(256) void f32_to_bf16_kernel(
    const float* __restrict__ in, unsigned short* __restrict__ out)
{
  const size_t base = ((size_t)blockIdx.x * 256 + threadIdx.x) * 16;
  const float4* s = (const float4*)(in + base);
  uint4* d = (uint4*)(out + base);
  d[0] = pack_bf16x8(s[0], s[1]);
  d[1] = pack_bf16x8(s[2], s[3]);
}

// C[M,N] = A[M,K] @ B[N,K]^T  (NT GEMM; bf16 operands, fp32 accumulate)
// Double-buffered LDS staged by global_load_async_to_lds_b128 (ASYNCcnt).
// OUT_BF16:    output bf16 (else fp32)
// TRANS_STORE: store C transposed (C[n][m]) -- used to build V^T
// MASK_SCALE:  fp32 epilogue: keep if col >= row (triu), scale, else -inf;
//              blocks entirely below the diagonal skip all compute.
// TRI_SKIP:    A[i,k] == 0 for k < i (softmax probs): start K loop at blockM.
template<bool OUT_BF16, bool TRANS_STORE, bool MASK_SCALE, bool TRI_SKIP>
__global__ __launch_bounds__(256, 2) void gemm_nt_kernel(
    const unsigned short* __restrict__ A, const unsigned short* __restrict__ B,
    void* __restrict__ Cv, float scale)
{
  __shared__ __bf16 As[2][BM * LDS_STRIDE];
  __shared__ __bf16 Bs[2][BN * LDS_STRIDE];

  const int tid  = threadIdx.x;
  const int lane = tid & 31;
  const int half = lane >> 4;   // 0: lanes 0-15, 1: lanes 16-31
  const int lm   = lane & 15;
  const int wid  = tid >> 5;    // 8 waves
  const int wm   = (wid >> 2) * 64;  // wave row base (2 waves along M)
  const int wn   = (wid & 3) * 32;   // wave col base (4 waves along N)

  const int blockM = blockIdx.y * BM;
  const int blockN = blockIdx.x * BN;

  // staging: each thread owns one 32-element (64B) half-row of the 128x64 tile
  const int lrow = tid >> 1;
  const int lcol = (tid & 1) * 32;

  // Entirely-masked score blocks (max col < min row): store -inf, no compute.
  if (MASK_SCALE && blockN + BN <= blockM) {
    float* C = (float*)Cv;
    const float ninf = -__builtin_inff();
    const float4 f4 = {ninf, ninf, ninf, ninf};
    float4* dst = (float4*)(C + (size_t)(blockM + lrow) * NDIM + blockN + (tid & 1) * 64);
#pragma unroll
    for (int j = 0; j < 16; ++j) dst[j] = f4;
    return;
  }

  // P@V: K tiles with k0+BK <= blockM are all-zero rows of P for this block.
  const int kbeg   = TRI_SKIP ? blockM : 0;       // blockM is a multiple of BK
  const int ksteps = (NDIM - kbeg) / BK;

  const unsigned short* ga = A + (size_t)(blockM + lrow) * NDIM + kbeg + lcol;
  const unsigned short* gb = B + (size_t)(blockN + lrow) * NDIM + kbeg + lcol;
  __bf16* la[2] = { &As[0][lrow * LDS_STRIDE + lcol], &As[1][lrow * LDS_STRIDE + lcol] };
  __bf16* lb[2] = { &Bs[0][lrow * LDS_STRIDE + lcol], &Bs[1][lrow * LDS_STRIDE + lcol] };

  v8f acc[4][2] = {};

  // prologue: stage 0 into buffer 0 (8 async-load instructions per wave)
  async_copy_row64B(ga, la[0]);
  async_copy_row64B(gb, lb[0]);

  for (int i = 0; i < ksteps; ++i) {
    const int par = i & 1;
    if (i + 1 < ksteps) {
      // issue stage i+1 into the other buffer, then wait for stage i's 8 loads
      async_copy_row64B(ga + (i + 1) * BK, la[par ^ 1]);
      async_copy_row64B(gb + (i + 1) * BK, lb[par ^ 1]);
      asm volatile("s_wait_asynccnt 0x8" ::: "memory");
    } else {
      asm volatile("s_wait_asynccnt 0x0" ::: "memory");
    }
    __syncthreads();   // every wave has waited for its own stage-i loads

    const __bf16* as = As[par];
    const __bf16* bs = Bs[par];
#pragma unroll
    for (int ks = 0; ks < BK; ks += 32) {
      v16bf afrag[4];
      v16bf bfrag[2];
#pragma unroll
      for (int tm = 0; tm < 4; ++tm) {
        // A 16x32 bf16 layout: lanes 0-15 K=[0..7]+[16..23], lanes 16-31 K=[8..15]+[24..31]
        const __bf16* p = as + (wm + tm * 16 + lm) * LDS_STRIDE + ks + half * 8;
        v8bf lo = *(const v8bf*)p;
        v8bf hi = *(const v8bf*)(p + 16);
        afrag[tm] = __builtin_shufflevector(lo, hi, 0,1,2,3,4,5,6,7,8,9,10,11,12,13,14,15);
      }
#pragma unroll
      for (int tn = 0; tn < 2; ++tn) {
        // B 32x16 bf16 layout: lanes 0-15 K=[0..7]+[16..23], lanes 16-31 K=[8..15]+[24..31]
        const __bf16* p = bs + (wn + tn * 16 + lm) * LDS_STRIDE + ks + half * 8;
        v8bf lo = *(const v8bf*)p;
        v8bf hi = *(const v8bf*)(p + 16);
        bfrag[tn] = __builtin_shufflevector(lo, hi, 0,1,2,3,4,5,6,7,8,9,10,11,12,13,14,15);
      }
#pragma unroll
      for (int tm = 0; tm < 4; ++tm)
#pragma unroll
        for (int tn = 0; tn < 2; ++tn)
          acc[tm][tn] = __builtin_amdgcn_wmma_f32_16x16x32_bf16(
              false, afrag[tm], false, bfrag[tn],
              (short)0, acc[tm][tn], false, false);
    }
    __syncthreads();   // all waves done reading buf[par] before it is refilled
  }

  // ---- epilogue: C/D layout: lane lm = column, VGPR r -> row (half*8 + r) ----
#pragma unroll
  for (int tm = 0; tm < 4; ++tm) {
#pragma unroll
    for (int tn = 0; tn < 2; ++tn) {
      const int gm = blockM + wm + tm * 16 + half * 8;  // first of 8 rows
      const int gn = blockN + wn + tn * 16 + lm;        // column
      if (OUT_BF16) {
        unsigned short* C = (unsigned short*)Cv;
        if (TRANS_STORE) {
          // C[gn][gm..gm+7]: contiguous -> one packed 16B store
          uint4 pk;
          pk.x = f32_to_bf16(acc[tm][tn][0]) | (f32_to_bf16(acc[tm][tn][1]) << 16);
          pk.y = f32_to_bf16(acc[tm][tn][2]) | (f32_to_bf16(acc[tm][tn][3]) << 16);
          pk.z = f32_to_bf16(acc[tm][tn][4]) | (f32_to_bf16(acc[tm][tn][5]) << 16);
          pk.w = f32_to_bf16(acc[tm][tn][6]) | (f32_to_bf16(acc[tm][tn][7]) << 16);
          *(uint4*)(C + (size_t)gn * NDIM + gm) = pk;
        } else {
#pragma unroll
          for (int r = 0; r < 8; ++r)
            C[(size_t)(gm + r) * NDIM + gn] = (unsigned short)f32_to_bf16(acc[tm][tn][r]);
        }
      } else {
        float* C = (float*)Cv;
#pragma unroll
        for (int r = 0; r < 8; ++r) {
          float v = acc[tm][tn][r];
          if (MASK_SCALE) {
            // reference keeps upper triangle: col >= row; others -> -inf; then /sqrt(N)
            v = (gn >= gm + r) ? v * scale : -__builtin_inff();
          }
          C[(size_t)(gm + r) * NDIM + gn] = v;
        }
      }
    }
  }
}

// Row softmax over S (fp32, with -inf masked entries) -> P (bf16)
__global__ __launch_bounds__(256) void softmax_rows_kernel(
    const float* __restrict__ S, unsigned short* __restrict__ P)
{
  __shared__ float red[256];
  const int row = blockIdx.x;
  const int tid = threadIdx.x;
  const float* s = S + (size_t)row * NDIM + tid * 16;

  float vals[16];
  const float4* s4 = (const float4*)s;
  float lmax = -__builtin_inff();
#pragma unroll
  for (int i = 0; i < 4; ++i) {
    float4 f = s4[i];
    vals[4 * i + 0] = f.x; vals[4 * i + 1] = f.y;
    vals[4 * i + 2] = f.z; vals[4 * i + 3] = f.w;
    lmax = fmaxf(lmax, fmaxf(fmaxf(f.x, f.y), fmaxf(f.z, f.w)));
  }
  red[tid] = lmax;
  __syncthreads();
  for (int off = 128; off > 0; off >>= 1) {
    if (tid < off) red[tid] = fmaxf(red[tid], red[tid + off]);
    __syncthreads();
  }
  const float rmax = red[0];
  __syncthreads();

  float e[16];
  float lsum = 0.0f;
#pragma unroll
  for (int i = 0; i < 16; ++i) {
    e[i] = __expf(vals[i] - rmax);   // exp(-inf) = 0 for masked entries
    lsum += e[i];
  }
  red[tid] = lsum;
  __syncthreads();
  for (int off = 128; off > 0; off >>= 1) {
    if (tid < off) red[tid] += red[tid + off];
    __syncthreads();
  }
  const float inv = 1.0f / red[0];

  uint4* p = (uint4*)(P + (size_t)row * NDIM + tid * 16);
#pragma unroll
  for (int i = 0; i < 2; ++i) {
    uint4 pk;
    pk.x = f32_to_bf16(e[8 * i + 0] * inv) | (f32_to_bf16(e[8 * i + 1] * inv) << 16);
    pk.y = f32_to_bf16(e[8 * i + 2] * inv) | (f32_to_bf16(e[8 * i + 3] * inv) << 16);
    pk.z = f32_to_bf16(e[8 * i + 4] * inv) | (f32_to_bf16(e[8 * i + 5] * inv) << 16);
    pk.w = f32_to_bf16(e[8 * i + 6] * inv) | (f32_to_bf16(e[8 * i + 7] * inv) << 16);
    p[i] = pk;
  }
}

extern "C" void kernel_launch(void* const* d_in, const int* in_sizes, int n_in,
                              void* d_out, int out_size, void* d_ws, size_t ws_size,
                              hipStream_t stream) {
  const float* x  = (const float*)d_in[0];
  const float* Wq = (const float*)d_in[1];
  const float* Wk = (const float*)d_in[2];
  const float* Wv = (const float*)d_in[3];
  float* out = (float*)d_out;

  const size_t NN = (size_t)NDIM * NDIM;
  // bf16 buffers are NN*2 bytes (32MB) each; lifetimes allow aliasing:
  //   S  (fp32, 64MB) overlays Wqb+Wkb   (dead after GEMMs 1-2)
  //   P  (bf16, 32MB) overlays Wvb       (dead after GEMM 3)
  unsigned short* xb  = (unsigned short*)d_ws;   // x   bf16, live to GEMM3
  unsigned short* Wqb = xb  + NN;
  unsigned short* Wkb = Wqb + NN;
  unsigned short* Wvb = Wkb + NN;
  unsigned short* Qb  = Wvb + NN;                // Q bf16
  unsigned short* Kb  = Qb  + NN;                // K bf16
  unsigned short* Vt  = Kb  + NN;                // V^T bf16
  float*          S   = (float*)Wqb;             // masked scores fp32 (aliases Wqb+Wkb)
  unsigned short* P   = Wvb;                     // softmax probs bf16 (aliases Wvb)

  dim3 grid(NDIM / BN, NDIM / BM);
  dim3 block(256);
  dim3 cgrid(NN / (256 * 16));
  const float scale = 1.0f / 64.0f;   // 1/sqrt(4096)

  // one-time fp32 -> bf16 conversion of all operands
  f32_to_bf16_kernel<<<cgrid, block, 0, stream>>>(x,  xb);
  f32_to_bf16_kernel<<<cgrid, block, 0, stream>>>(Wq, Wqb);
  f32_to_bf16_kernel<<<cgrid, block, 0, stream>>>(Wk, Wkb);
  f32_to_bf16_kernel<<<cgrid, block, 0, stream>>>(Wv, Wvb);

  // projections: Q = x@Wq^T, K = x@Wk^T, V^T = (x@Wv^T)^T
  gemm_nt_kernel<true, false, false, false><<<grid, block, 0, stream>>>(xb, Wqb, Qb, 1.0f);
  gemm_nt_kernel<true, false, false, false><<<grid, block, 0, stream>>>(xb, Wkb, Kb, 1.0f);
  gemm_nt_kernel<true, true,  false, false><<<grid, block, 0, stream>>>(xb, Wvb, Vt, 1.0f);
  // scores: S = mask(Q@K^T) / 64; blocks fully below diagonal store -inf directly
  gemm_nt_kernel<false, false, true, false><<<grid, block, 0, stream>>>(Qb, Kb, S, scale);
  // row softmax -> bf16 P  (P overlays Wvb, dead now)
  softmax_rows_kernel<<<dim3(NDIM), block, 0, stream>>>(S, P);
  // out = P @ V: P rows are zero for k < i, so K loop starts at blockM
  gemm_nt_kernel<false, false, false, true><<<grid, block, 0, stream>>>(P, Vt, out, 1.0f);
}
